// QuantumInspiredLLM_12137577578772
// MI455X (gfx1250) — matrix operand order repaired and checked by reading, not compiled, not run on hardware
//
#include <hip/hip_runtime.h>
#include <hip/hip_bf16.h>
#include <math.h>

// Model constants (QuantumInspiredLLM): V=32000 D=1024 DEPTH=6 H=16 L=1024 B=2 R=16 HID=4096
// SCALING = 8/16 = 0.5, HD = 64.

typedef __attribute__((ext_vector_type(16))) __bf16 v16bf;
typedef __attribute__((ext_vector_type(8)))  __bf16 v8bf;
typedef __attribute__((ext_vector_type(8)))  float  v8f;

// ---------------------------------------------------------------------------
// Packed-bf16 WMMA fragment loaders.
// A 16x32 (MxK) 16-bit layout: lane -> m = lane&15, hf = lane>>4
//   vector elems 0..7  = K (hf*16 .. hf*16+7)      -> contiguous 16B
//   vector elems 8..15 = K (hf*16+8 .. hf*16+15)   -> contiguous 16B
// ---------------------------------------------------------------------------
__device__ __forceinline__ v16bf load_a_bf(const __bf16* __restrict__ base,
                                           int ld, int lane) {
  const int m = lane & 15, hf = lane >> 4;
  const __bf16* row = base + (size_t)m * ld;
  const v8bf lo = *(const v8bf*)(row + hf * 16);       // global_load_b128
  const v8bf hi = *(const v8bf*)(row + 8 + hf * 16);   // global_load_b128
  v16bf r;
#pragma unroll
  for (int i = 0; i < 8; ++i) { r[i] = lo[i]; r[8 + i] = hi[i]; }
  return r;
}

// B 32x16 (KxN) from a [N,K]-style row-major source (B[k][n] = base[n*ld+k]):
// lane -> n = lane&15, hf = lane>>4 ; elems 0..15 = K (hf*16 .. hf*16+15)
__device__ __forceinline__ v16bf load_b_bf(const __bf16* __restrict__ base,
                                           int ld, int lane) {
  const int n = lane & 15, hf = lane >> 4;
  const __bf16* row = base + (size_t)n * ld;
  return *(const v16bf*)(row + hf * 16);               // 2x global_load_b128
}

__device__ __forceinline__ v8f wmma_bf16(const v16bf& a, const v16bf& b,
                                         const v8f& c) {
  return __builtin_amdgcn_wmma_f32_16x16x32_bf16(false, a, false, b,
                                                 (short)0, c, false, false);
}

// ---------------------------------------------------------------------------
// GEMM: C[M,N] = act(A[M,K] @ W[N,K]^T + bias) (+ residual)
// A, W packed bf16; accum f32. 256 thr = 8 waves arranged 2(M) x 4(N);
// each wave computes a 32x32 tile (4 WMMAs / 8 b128 loads per K-step),
// software-pipelined one K-step ahead. Block tile 64(M) x 128(N).
// act: 0 none, 1 exact GELU. If Cbf != nullptr, write bf16 there instead of C.
// ---------------------------------------------------------------------------
__global__ __launch_bounds__(256) void gemm_bf16_wmma(
    const __bf16* __restrict__ A, const __bf16* __restrict__ W,
    const float* __restrict__ bias, const float* __restrict__ residual,
    float* __restrict__ C, __bf16* __restrict__ Cbf,
    int M, int N, int K, int act) {
  const int lane = threadIdx.x & 31;
  const int wave = threadIdx.x >> 5;
  const int tm = blockIdx.y * 64 + (wave >> 2) * 32;
  const int tn = blockIdx.x * 128 + (wave & 3) * 32;
  const __bf16* Ap0 = A + (size_t)tm * K;
  const __bf16* Ap1 = Ap0 + (size_t)16 * K;
  const __bf16* Wp0 = W + (size_t)tn * K;
  const __bf16* Wp1 = Wp0 + (size_t)16 * K;

  v8f acc00 = {0.f, 0.f, 0.f, 0.f, 0.f, 0.f, 0.f, 0.f};
  v8f acc01 = acc00, acc10 = acc00, acc11 = acc00;

  // prologue: fragments for K-step 0
  v16bf a0 = load_a_bf(Ap0, K, lane);
  v16bf a1 = load_a_bf(Ap1, K, lane);
  v16bf b0 = load_b_bf(Wp0, K, lane);
  v16bf b1 = load_b_bf(Wp1, K, lane);

  for (int k0 = 32; k0 < K; k0 += 32) {
    // issue next K-step's loads before the current WMMAs (latency overlap)
    __builtin_prefetch(Ap0 + k0 + 32, 0, 3);
    __builtin_prefetch(Wp0 + k0 + 32, 0, 3);
    v16bf na0 = load_a_bf(Ap0 + k0, K, lane);
    v16bf na1 = load_a_bf(Ap1 + k0, K, lane);
    v16bf nb0 = load_b_bf(Wp0 + k0, K, lane);
    v16bf nb1 = load_b_bf(Wp1 + k0, K, lane);
    acc00 = wmma_bf16(a0, b0, acc00);
    acc01 = wmma_bf16(a0, b1, acc01);
    acc10 = wmma_bf16(a1, b0, acc10);
    acc11 = wmma_bf16(a1, b1, acc11);
    a0 = na0; a1 = na1; b0 = nb0; b1 = nb1;
  }
  acc00 = wmma_bf16(a0, b0, acc00);
  acc01 = wmma_bf16(a0, b1, acc01);
  acc10 = wmma_bf16(a1, b0, acc10);
  acc11 = wmma_bf16(a1, b1, acc11);

  const int hf = lane >> 4, ln = lane & 15;
  auto store_tile = [&](const v8f& acc, int m0, int n0) {
    const int n = n0 + ln;
    const float bv = bias ? bias[n] : 0.0f;
#pragma unroll
    for (int i = 0; i < 8; ++i) {
      const int m = m0 + hf * 8 + i;
      float y = acc[i] + bv;
      if (act == 1) y = 0.5f * y * (1.0f + erff(y * 0.70710678118654752f));
      if (Cbf) {
        Cbf[(size_t)m * N + n] = (__bf16)y;
      } else {
        if (residual) y += residual[(size_t)m * N + n];
        C[(size_t)m * N + n] = y;
      }
    }
  };
  store_tile(acc00, tm, tn);
  store_tile(acc01, tm, tn + 16);
  store_tile(acc10, tm + 16, tn);
  store_tile(acc11, tm + 16, tn + 16);
}

// ---------------------------------------------------------------------------
// Fused causal attention per (b, h, 16-query tile). L=1024, HD=64, D=1024.
// q,k: bf16 [B*L, D]. vt: bf16 head-transposed [B,H,HD,L]. o: bf16 [B*L, D].
// 128 threads = 4 waves. LDS: 64 KB f32 scores + 32 KB bf16 P (WGP has 320 KB).
// ---------------------------------------------------------------------------
__global__ __launch_bounds__(128) void attn_wmma(
    const __bf16* __restrict__ q, const __bf16* __restrict__ k,
    const __bf16* __restrict__ vt, __bf16* __restrict__ o) {
  constexpr int Lc = 1024, Dc = 1024, HDc = 64, Hc = 16;
  __shared__ float  S[16][Lc];
  __shared__ __bf16 P[16][Lc];
  const int lane = threadIdx.x & 31;
  const int wave = threadIdx.x >> 5;
  const int hf = lane >> 4;
  const int qbase = blockIdx.x * 16;
  const int hh = blockIdx.y;
  const size_t rowbase = (size_t)blockIdx.z * Lc;

  // ---- stage 1: S = mask(QK^T * scale). Each wave owns key tiles kt = wave*16 + 64t.
  const __bf16* qp = q + (rowbase + qbase) * Dc + hh * HDc;
  const v16bf qa0 = load_a_bf(qp, Dc, lane);       // head dims 0..31
  const v16bf qa1 = load_a_bf(qp + 32, Dc, lane);  // head dims 32..63
  const float scale = 0.125f;                      // 64^-0.5
  for (int kt = wave * 16; kt < Lc; kt += 64) {
    const __bf16* kp = k + (rowbase + kt) * Dc + hh * HDc;
    v16bf kb0 = load_b_bf(kp, Dc, lane);
    v16bf kb1 = load_b_bf(kp + 32, Dc, lane);
    v8f acc = {0.f, 0.f, 0.f, 0.f, 0.f, 0.f, 0.f, 0.f};
    acc = wmma_bf16(qa0, kb0, acc);
    acc = wmma_bf16(qa1, kb1, acc);
    const int key = kt + (lane & 15);
#pragma unroll
    for (int i = 0; i < 8; ++i) {
      const int r = hf * 8 + i;
      S[r][key] = (key <= qbase + r) ? acc[i] * scale : -INFINITY;
    }
  }
  __syncthreads();

  // ---- stage 2: row softmax (f32 in LDS), write normalized P as bf16.
  for (int j = 0; j < 4; ++j) {
    const int row = wave * 4 + j;
    float mx = -INFINITY;
    for (int c = lane; c < Lc; c += 32) mx = fmaxf(mx, S[row][c]);
#pragma unroll
    for (int off = 16; off > 0; off >>= 1) mx = fmaxf(mx, __shfl_xor(mx, off, 32));
    float sum = 0.f;
    for (int c = lane; c < Lc; c += 32) {
      const float e = __expf(S[row][c] - mx);
      S[row][c] = e;
      sum += e;
    }
#pragma unroll
    for (int off = 16; off > 0; off >>= 1) sum += __shfl_xor(sum, off, 32);
    const float inv = 1.0f / sum;
    for (int c = lane; c < Lc; c += 32) P[row][c] = (__bf16)(S[row][c] * inv);
  }
  __syncthreads();

  // ---- stage 3: O = P @ V^T-layout. Each wave owns a 16-wide slice of HD=64.
  v8f acc = {0.f, 0.f, 0.f, 0.f, 0.f, 0.f, 0.f, 0.f};
  const int c0 = wave * 16;  // head-dim slice
  const __bf16* vtp = vt + (((size_t)blockIdx.z * Hc + hh) * HDc + c0) * Lc;
  for (int kt = 0; kt < Lc; kt += 32) {
    v16bf pa = load_a_bf(&P[0][kt], Lc, lane);   // ds_load_b128 from LDS
    v16bf vb = load_b_bf(vtp + kt, Lc, lane);    // contiguous keys per head-col
    acc = wmma_bf16(pa, vb, acc);
  }
  const int n = lane & 15;
#pragma unroll
  for (int i = 0; i < 8; ++i) {
    const int m = hf * 8 + i;
    o[(rowbase + qbase + m) * Dc + hh * HDc + c0 + n] = (__bf16)acc[i];
  }
}

// ---------------------------------------------------------------------------
// Embedding + PhaseRotator: x = (tok[idx] + pos) * (cos(phi) - sin(phi))  (f32)
// ---------------------------------------------------------------------------
__global__ __launch_bounds__(256) void embed_kernel(
    const int* __restrict__ idx, const float* __restrict__ tok,
    const float* __restrict__ pos, const float* __restrict__ phase,
    float* __restrict__ x) {
  constexpr int Dc = 1024, Lc = 1024;
  const int bl = blockIdx.x;
  const int l = bl & (Lc - 1);
  const int t = idx[bl];
  for (int d = threadIdx.x; d < Dc; d += 256) {
    const float ph = tanhf(phase[d]) * 3.14159265358979323846f;
    x[(size_t)bl * Dc + d] =
        (tok[(size_t)t * Dc + d] + pos[l * Dc + d]) * (__cosf(ph) - __sinf(ph));
  }
}

// ---------------------------------------------------------------------------
// LayerNorm over D=1024, one row per block; f32 in, packed bf16 out (GEMM A).
// ---------------------------------------------------------------------------
__global__ __launch_bounds__(256) void layernorm_bf_kernel(
    const float* __restrict__ x, const float* __restrict__ w,
    const float* __restrict__ b, __bf16* __restrict__ y) {
  constexpr int Dc = 1024;
  __shared__ float red[8];
  const int tid = threadIdx.x, lane = tid & 31, wave = tid >> 5;
  const float* xr = x + (size_t)blockIdx.x * Dc;
  float s = 0.f;
  for (int i = tid; i < Dc; i += 256) s += xr[i];
#pragma unroll
  for (int off = 16; off > 0; off >>= 1) s += __shfl_xor(s, off, 32);
  if (lane == 0) red[wave] = s;
  __syncthreads();
  float tot = 0.f;
#pragma unroll
  for (int i = 0; i < 8; ++i) tot += red[i];
  const float mean = tot * (1.0f / Dc);
  __syncthreads();
  float vs = 0.f;
  for (int i = tid; i < Dc; i += 256) {
    const float d = xr[i] - mean;
    vs += d * d;
  }
#pragma unroll
  for (int off = 16; off > 0; off >>= 1) vs += __shfl_xor(vs, off, 32);
  if (lane == 0) red[wave] = vs;
  __syncthreads();
  float vt = 0.f;
#pragma unroll
  for (int i = 0; i < 8; ++i) vt += red[i];
  const float rstd = rsqrtf(vt * (1.0f / Dc) + 1e-5f);
  for (int i = tid; i < Dc; i += 256)
    y[(size_t)blockIdx.x * Dc + i] = (__bf16)((xr[i] - mean) * rstd * w[i] + b[i]);
}

// ---------------------------------------------------------------------------
// f32 -> packed bf16 (vectorized, grid-stride); n must be a multiple of 4.
// ---------------------------------------------------------------------------
__global__ __launch_bounds__(256) void cvt_bf16_kernel(
    const float* __restrict__ in, __bf16* __restrict__ out, long n) {
  const long stride = (long)gridDim.x * 1024;
  for (long i = (long)blockIdx.x * 1024 + (long)threadIdx.x * 4; i < n; i += stride) {
    const float4 f = *(const float4*)(in + i);
    out[i + 0] = (__bf16)f.x;
    out[i + 1] = (__bf16)f.y;
    out[i + 2] = (__bf16)f.z;
    out[i + 3] = (__bf16)f.w;
  }
}

// V f32 [B*L, D] -> bf16 head-transposed [B, H, HD, L]
__global__ __launch_bounds__(256) void vt_cvt_kernel(
    const float* __restrict__ v, __bf16* __restrict__ vt) {
  constexpr int Dc = 1024, Lc = 1024, Hc = 16, HDc = 64;
  const int gid = blockIdx.x * 256 + threadIdx.x;  // over M*D = 2^21
  const int row = gid >> 10, d = gid & (Dc - 1);
  const int b = row >> 10, l = row & (Lc - 1);
  const int h = d >> 6, c = d & (HDc - 1);
  vt[(((size_t)b * Hc + h) * HDc + c) * Lc + l] = (__bf16)v[gid];
}

// ---------------------------------------------------------------------------
// LoRA: mid[M,16] = h[M,1024] @ Bm[16,1024]^T   (0.2% of FLOPs -> VALU ok)
// ---------------------------------------------------------------------------
__global__ __launch_bounds__(256) void lora_mid_kernel(
    const __bf16* __restrict__ h, const float* __restrict__ Bm,
    float* __restrict__ mid) {
  constexpr int Kc = 1024, Rc = 16;
  const int gid = blockIdx.x * 256 + threadIdx.x;
  const int m = gid >> 4, r = gid & (Rc - 1);
  const __bf16* hr = h + (size_t)m * Kc;
  const float* br = Bm + (size_t)r * Kc;
  float s = 0.f;
  for (int k2 = 0; k2 < Kc; ++k2) s = fmaf((float)hr[k2], br[k2], s);
  mid[gid] = s;
}

// y[M,1024] += 0.5 * mid[M,16] @ Am[1024,16]^T
__global__ __launch_bounds__(256) void lora_apply_kernel(
    const float* __restrict__ mid, const float* __restrict__ Am,
    float* __restrict__ y) {
  constexpr int Nc = 1024, Rc = 16;
  const int gid = blockIdx.x * 256 + threadIdx.x;
  const int m = gid >> 10, n = gid & (Nc - 1);
  float s = 0.f;
#pragma unroll
  for (int r = 0; r < Rc; ++r) s = fmaf(mid[m * Rc + r], Am[n * Rc + r], s);
  y[gid] += 0.5f * s;  // SCALING = 8/16
}

// ---------------------------------------------------------------------------
extern "C" void kernel_launch(void* const* d_in, const int* in_sizes, int n_in,
                              void* d_out, int out_size, void* d_ws,
                              size_t ws_size, hipStream_t stream) {
  (void)in_sizes; (void)n_in; (void)out_size; (void)ws_size;
  constexpr int Bc = 2, Lc = 1024, Dc = 1024, DEPTH = 6, Hc = 16, HID = 4096,
                Vc = 32000, Rc = 16;
  constexpr int M = Bc * Lc;  // 2048 rows

  const int*   idx   = (const int*)d_in[0];
  const float* tok   = (const float*)d_in[1];
  const float* pos   = (const float*)d_in[2];
  const float* phase = (const float*)d_in[3];
  const float* ln1w  = (const float*)d_in[4];
  const float* ln1b  = (const float*)d_in[5];
  const float* ln2w  = (const float*)d_in[6];
  const float* ln2b  = (const float*)d_in[7];
  const float* Wq = (const float*)d_in[8];  const float* bq = (const float*)d_in[9];
  const float* Bq = (const float*)d_in[10]; const float* Aq = (const float*)d_in[11];
  const float* Wk = (const float*)d_in[12]; const float* bk = (const float*)d_in[13];
  const float* Bk = (const float*)d_in[14]; const float* Ak = (const float*)d_in[15];
  const float* Wv = (const float*)d_in[16]; const float* bv = (const float*)d_in[17];
  const float* Bv = (const float*)d_in[18]; const float* Av = (const float*)d_in[19];
  const float* Wo = (const float*)d_in[20]; const float* bo = (const float*)d_in[21];
  const float* Bo = (const float*)d_in[22]; const float* Ao = (const float*)d_in[23];
  const float* W1 = (const float*)d_in[24]; const float* b1 = (const float*)d_in[25];
  const float* W2 = (const float*)d_in[26]; const float* b2 = (const float*)d_in[27];
  const float* lnfw  = (const float*)d_in[28];
  const float* lnfb  = (const float*)d_in[29];
  const float* headW = (const float*)d_in[30];

  // ---- workspace layout (~142 MB) ----
  char* ws = (char*)d_ws;
  const size_t MD4 = (size_t)M * Dc * sizeof(float);    // 8 MB
  const size_t MD2 = (size_t)M * Dc * sizeof(__bf16);   // 4 MB
  size_t off = 0;
  float* xA  = (float*)(ws + off); off += MD4;
  float* xB  = (float*)(ws + off); off += MD4;
  float* qb  = (float*)(ws + off); off += MD4;
  float* kb  = (float*)(ws + off); off += MD4;
  float* vb  = (float*)(ws + off); off += MD4;
  float* mid = (float*)(ws + off); off += (size_t)M * Rc * sizeof(float);
  __bf16* hbf   = (__bf16*)(ws + off); off += MD2;
  __bf16* qbf   = (__bf16*)(ws + off); off += MD2;
  __bf16* kbf   = (__bf16*)(ws + off); off += MD2;
  __bf16* vtbf  = (__bf16*)(ws + off); off += MD2;
  __bf16* obf   = (__bf16*)(ws + off); off += MD2;
  __bf16* hidbf = (__bf16*)(ws + off); off += (size_t)M * HID * sizeof(__bf16);   // 16 MB
  __bf16* wbf   = (__bf16*)(ws + off); off += (size_t)Vc * Dc * sizeof(__bf16);   // 65.5 MB

  const dim3 blk256(256), blk128(128);
  const dim3 gD(Dc / 128, M / 64);     // N=1024 GEMMs
  const dim3 gHID(HID / 128, M / 64);  // N=4096 GEMM
  const dim3 gHead(Vc / 128, M / 64);  // N=32000 GEMM
  const dim3 gAttn(Lc / 16, Hc, Bc);
  const long nDD = (long)Dc * Dc, nHD = (long)HID * Dc;
  const int gLora = M * Rc / 256, gMD = M * Dc / 256;

  embed_kernel<<<M, blk256, 0, stream>>>(idx, tok, pos, phase, xA);

  for (int i = 0; i < DEPTH; ++i) {
    const size_t wOff = (size_t)i * Dc * Dc;
    // --- attention block ---
    layernorm_bf_kernel<<<M, blk256, 0, stream>>>(xA, ln1w + i * Dc, ln1b + i * Dc, hbf);

    cvt_bf16_kernel<<<1024, blk256, 0, stream>>>(Wq + wOff, wbf, nDD);
    gemm_bf16_wmma<<<gD, blk256, 0, stream>>>(hbf, wbf, bq + i * Dc, nullptr,
                                              qb, nullptr, M, Dc, Dc, 0);
    lora_mid_kernel<<<gLora, blk256, 0, stream>>>(hbf, Bq + (size_t)i * Rc * Dc, mid);
    lora_apply_kernel<<<gMD, blk256, 0, stream>>>(mid, Aq + (size_t)i * Dc * Rc, qb);
    cvt_bf16_kernel<<<1024, blk256, 0, stream>>>(qb, qbf, (long)M * Dc);

    cvt_bf16_kernel<<<1024, blk256, 0, stream>>>(Wk + wOff, wbf, nDD);
    gemm_bf16_wmma<<<gD, blk256, 0, stream>>>(hbf, wbf, bk + i * Dc, nullptr,
                                              kb, nullptr, M, Dc, Dc, 0);
    lora_mid_kernel<<<gLora, blk256, 0, stream>>>(hbf, Bk + (size_t)i * Rc * Dc, mid);
    lora_apply_kernel<<<gMD, blk256, 0, stream>>>(mid, Ak + (size_t)i * Dc * Rc, kb);
    cvt_bf16_kernel<<<1024, blk256, 0, stream>>>(kb, kbf, (long)M * Dc);

    cvt_bf16_kernel<<<1024, blk256, 0, stream>>>(Wv + wOff, wbf, nDD);
    gemm_bf16_wmma<<<gD, blk256, 0, stream>>>(hbf, wbf, bv + i * Dc, nullptr,
                                              vb, nullptr, M, Dc, Dc, 0);
    lora_mid_kernel<<<gLora, blk256, 0, stream>>>(hbf, Bv + (size_t)i * Rc * Dc, mid);
    lora_apply_kernel<<<gMD, blk256, 0, stream>>>(mid, Av + (size_t)i * Dc * Rc, vb);
    vt_cvt_kernel<<<gMD, blk256, 0, stream>>>(vb, vtbf);

    attn_wmma<<<gAttn, blk128, 0, stream>>>(qbf, kbf, vtbf, obf);

    cvt_bf16_kernel<<<1024, blk256, 0, stream>>>(Wo + wOff, wbf, nDD);
    gemm_bf16_wmma<<<gD, blk256, 0, stream>>>(obf, wbf, bo + i * Dc, xA,
                                              xB, nullptr, M, Dc, Dc, 0);
    lora_mid_kernel<<<gLora, blk256, 0, stream>>>(obf, Bo + (size_t)i * Rc * Dc, mid);
    lora_apply_kernel<<<gMD, blk256, 0, stream>>>(mid, Ao + (size_t)i * Dc * Rc, xB);

    // --- MLP block ---
    layernorm_bf_kernel<<<M, blk256, 0, stream>>>(xB, ln2w + i * Dc, ln2b + i * Dc, hbf);
    cvt_bf16_kernel<<<4096, blk256, 0, stream>>>(W1 + (size_t)i * HID * Dc, wbf, nHD);
    gemm_bf16_wmma<<<gHID, blk256, 0, stream>>>(hbf, wbf, b1 + i * HID, nullptr,
                                                nullptr, hidbf, M, HID, Dc, 1);
    cvt_bf16_kernel<<<4096, blk256, 0, stream>>>(W2 + (size_t)i * Dc * HID, wbf, nHD);
    gemm_bf16_wmma<<<gD, blk256, 0, stream>>>(hidbf, wbf, b2 + i * Dc, xB,
                                              xA, nullptr, M, Dc, HID, 0);
  }

  layernorm_bf_kernel<<<M, blk256, 0, stream>>>(xA, lnfw, lnfb, hbf);
  cvt_bf16_kernel<<<8192, blk256, 0, stream>>>(headW, wbf, (long)Vc * Dc);
  gemm_bf16_wmma<<<gHead, blk256, 0, stream>>>(hbf, wbf, nullptr, nullptr,
                                               (float*)d_out, nullptr, M, Vc, Dc, 0);
}
